// SelfAttention_16277926052531
// MI455X (gfx1250) — matrix operand order repaired and checked by reading, not compiled, not run on hardware
//
#include <hip/hip_runtime.h>
#include <hip/hip_bf16.h>

typedef __attribute__((ext_vector_type(16))) __bf16 v16bf;
typedef __attribute__((ext_vector_type(8)))  float  v8f;
typedef unsigned short u16;
typedef unsigned int   u32;

#define C_EMBD 1024
#define HEADS  16
#define HDIM   64
#define BATCH  2
#define TSEQ   2048
#define MROWS  (BATCH * TSEQ)   // 4096
#define NQKV   (3 * C_EMBD)     // 3072

union Frag {
  v16bf v;
  uint4 q[2];
};

__device__ __forceinline__ u16 f2bf(float f) {
  u32 u = __float_as_uint(f);
  u32 r = u + 0x7FFFu + ((u >> 16) & 1u);   // round-to-nearest-even
  return (u16)(r >> 16);
}

// A fragment: 16x32 (MxK) bf16, row-major source with leading dim ld.
// lane L: row m = L&15, kgroup = L>>4; elems 0..7 at k=8*kg, elems 8..15 at k=16+8*kg.
__device__ __forceinline__ v16bf load_fragA(const u16* base, int ld, int lane) {
  Frag f;
  const u16* p = base + (size_t)(lane & 15) * ld + 8 * (lane >> 4);
  f.q[0] = *(const uint4*)p;
  f.q[1] = *(const uint4*)(p + 16);
  return f.v;
}

// B fragment: 32x16 (KxN) bf16 from an N-major source (row n contiguous over k).
// lane L: col n = L&15, kgroup = L>>4; elems 0..7 at k=8*kg, elems 8..15 at k=16+8*kg.
__device__ __forceinline__ v16bf load_fragB(const u16* base, int ld, int lane) {
  Frag f;
  const u16* p = base + (size_t)(lane & 15) * ld + 8 * (lane >> 4);
  f.q[0] = *(const uint4*)p;
  f.q[1] = *(const uint4*)(p + 16);
  return f.v;
}

__device__ __forceinline__ v8f wmma_bf16(v16bf a, v16bf b, v8f c) {
  return __builtin_amdgcn_wmma_f32_16x16x32_bf16(false, a, false, b, (short)0, c,
                                                 false, false);
}

// ---------------- conversion kernels ----------------
__global__ void k_cvt(const float* __restrict__ in, u16* __restrict__ out, int n) {
  int i = blockIdx.x * blockDim.x + threadIdx.x;
  if (i < n) out[i] = f2bf(in[i]);
}

// in: KxN row-major fp32 -> out: NxK row-major bf16 (transpose)
__global__ void k_cvt_T(const float* __restrict__ in, u16* __restrict__ out,
                        int K, int N) {
  int i = blockIdx.x * blockDim.x + threadIdx.x;
  if (i >= K * N) return;
  int n = i / K, k = i % K;
  out[i] = f2bf(in[(size_t)k * N + n]);
}

// ---------------- QKV GEMM: [4096,1024] x [1024,3072] + bias -> Q/K/Vt bf16 ----
__global__ void __launch_bounds__(256) k_qkv(const u16* __restrict__ xb,
                                             const u16* __restrict__ waT,
                                             const float* __restrict__ b_attn,
                                             u16* __restrict__ Q,
                                             u16* __restrict__ Kk,
                                             u16* __restrict__ Vt) {
  int lane = threadIdx.x & 31;
  int wave = blockIdx.x * (blockDim.x >> 5) + (threadIdx.x >> 5);
  const int NT = NQKV / 32;                 // 96 n-tiles
  int mt = wave / NT, nt = wave % NT;
  int m0 = mt * 32, n0 = nt * 32;

  v8f acc[2][2] = {};
  for (int k0 = 0; k0 < C_EMBD; k0 += 32) {
    v16bf a0 = load_fragA(xb + (size_t)m0 * C_EMBD + k0, C_EMBD, lane);
    v16bf a1 = load_fragA(xb + (size_t)(m0 + 16) * C_EMBD + k0, C_EMBD, lane);
    v16bf b0 = load_fragB(waT + (size_t)n0 * C_EMBD + k0, C_EMBD, lane);
    v16bf b1 = load_fragB(waT + (size_t)(n0 + 16) * C_EMBD + k0, C_EMBD, lane);
    acc[0][0] = wmma_bf16(a0, b0, acc[0][0]);
    acc[0][1] = wmma_bf16(a0, b1, acc[0][1]);
    acc[1][0] = wmma_bf16(a1, b0, acc[1][0]);
    acc[1][1] = wmma_bf16(a1, b1, acc[1][1]);
  }

  for (int i = 0; i < 2; ++i)
    for (int j = 0; j < 2; ++j)
      for (int r = 0; r < 8; ++r) {
        int m = m0 + 16 * i + r + 8 * (lane >> 4);
        int n = n0 + 16 * j + (lane & 15);
        float val = acc[i][j][r] + b_attn[n];
        int t = m & (TSEQ - 1), bb = m >> 11;
        int sec = n >> 10, cin = n & (C_EMBD - 1);
        int h = cin >> 6, d = cin & 63;
        size_t bh = (size_t)bb * HEADS + h;
        u16 bv = f2bf(val);
        if (sec == 0)      Q [(bh * TSEQ + t) * HDIM + d] = bv;
        else if (sec == 1) Kk[(bh * TSEQ + t) * HDIM + d] = bv;
        else               Vt[(bh * HDIM + d) * TSEQ + t] = bv;
      }
}

// ---------------- flash attention: one wave per 16-query tile per (b,h) -------
__global__ void __launch_bounds__(256) k_attn(const u16* __restrict__ Q,
                                              const u16* __restrict__ Kk,
                                              const u16* __restrict__ Vt,
                                              u16* __restrict__ att) {
  __shared__ u16 ldsP[8][16 * 32];          // per-wave 16x32 bf16 P staging
  int lane  = threadIdx.x & 31;
  int wslot = threadIdx.x >> 5;
  int wave  = blockIdx.x * (blockDim.x >> 5) + wslot;
  int qt = wave & 127;                      // 128 query tiles of 16
  int bh = wave >> 7;                       // 0..31
  int q0 = qt * 16;

  const u16* Qbh = Q  + (size_t)bh * TSEQ * HDIM;
  const u16* Kbh = Kk + (size_t)bh * TSEQ * HDIM;
  const u16* Vbh = Vt + (size_t)bh * HDIM * TSEQ;

  v16bf aq0 = load_fragA(Qbh + (size_t)q0 * HDIM + 0,  HDIM, lane);
  v16bf aq1 = load_fragA(Qbh + (size_t)q0 * HDIM + 32, HDIM, lane);

  float rmax[8], rsum[8];
  v8f o[4] = {};
  for (int r = 0; r < 8; ++r) { rmax[r] = -1e30f; rsum[r] = 0.f; }

  const float scale = 0.125f;               // 1/sqrt(64)
  int ncol = lane & 15;
  int myq[8];
  for (int r = 0; r < 8; ++r) myq[r] = q0 + r + 8 * (lane >> 4);

  for (int kb = 0; kb < q0 + 16; kb += 32) {
    // --- S = Q @ K^T (two 16x16 tiles over a 32-key block) ---
    v8f s[2] = {};
    for (int j = 0; j < 2; ++j) {
      int kx = kb + 16 * j;
      v16bf b0 = load_fragB(Kbh + (size_t)kx * HDIM + 0,  HDIM, lane);
      v16bf b1 = load_fragB(Kbh + (size_t)kx * HDIM + 32, HDIM, lane);
      s[j] = wmma_bf16(aq0, b0, s[j]);
      s[j] = wmma_bf16(aq1, b1, s[j]);
    }
    // --- causal mask + scale + block row-max ---
    float sv[2][8], bmax[8];
    for (int r = 0; r < 8; ++r) bmax[r] = -1e30f;
    for (int j = 0; j < 2; ++j) {
      int kn = kb + 16 * j + ncol;
      for (int r = 0; r < 8; ++r) {
        float v = s[j][r] * scale;
        v = (kn <= myq[r]) ? v : -1e30f;
        sv[j][r] = v;
        bmax[r] = fmaxf(bmax[r], v);
      }
    }
    for (int m = 1; m < 16; m <<= 1)
      for (int r = 0; r < 8; ++r)
        bmax[r] = fmaxf(bmax[r], __shfl_xor(bmax[r], m, 32));
    // --- online softmax update ---
    float corr[8], bsum[8];
    for (int r = 0; r < 8; ++r) {
      float nm = fmaxf(rmax[r], bmax[r]);
      corr[r] = __expf(rmax[r] - nm);
      rmax[r] = nm;
      bsum[r] = 0.f;
    }
    for (int j = 0; j < 2; ++j)
      for (int r = 0; r < 8; ++r) {
        float p = __expf(sv[j][r] - rmax[r]);
        bsum[r] += p;
        int row = r + 8 * (lane >> 4);
        int col = 16 * j + ncol;
        ldsP[wslot][row * 32 + col] = f2bf(p);   // re-layout P via LDS
      }
    for (int m = 1; m < 16; m <<= 1)
      for (int r = 0; r < 8; ++r)
        bsum[r] += __shfl_xor(bsum[r], m, 32);
    for (int r = 0; r < 8; ++r)
      rsum[r] = rsum[r] * corr[r] + bsum[r];
    for (int t = 0; t < 4; ++t)
      for (int r = 0; r < 8; ++r)
        o[t][r] *= corr[r];
    // wave-local LDS RAW: wait for the 16 ds_store_b16 before ds_load_b128
    asm volatile("s_wait_dscnt 0" ::: "memory");
    v16bf pf = load_fragA(&ldsP[wslot][0], 32, lane);
    // --- O += P @ V  (Vt is d-major so B frags are contiguous) ---
    for (int t = 0; t < 4; ++t) {
      v16bf bv = load_fragB(Vbh + (size_t)(16 * t) * TSEQ + kb, TSEQ, lane);
      o[t] = wmma_bf16(pf, bv, o[t]);
    }
  }

  // normalize + write att in [M, C] bf16 for the proj GEMM
  int h  = bh & (HEADS - 1);
  int bb = bh >> 4;
  for (int r = 0; r < 8; ++r) {
    float inv = 1.0f / rsum[r];
    size_t m = (size_t)bb * TSEQ + q0 + r + 8 * (lane >> 4);
    for (int t = 0; t < 4; ++t) {
      int c = h * HDIM + 16 * t + ncol;
      att[m * C_EMBD + c] = f2bf(o[t][r] * inv);
    }
  }
}

// ---------------- proj GEMM: [4096,1024] x [1024,1024] + bias -> fp32 out -----
__global__ void __launch_bounds__(256) k_proj(const u16* __restrict__ att,
                                              const u16* __restrict__ wpT,
                                              const float* __restrict__ b_proj,
                                              float* __restrict__ out) {
  int lane = threadIdx.x & 31;
  int wave = blockIdx.x * (blockDim.x >> 5) + (threadIdx.x >> 5);
  const int NT = C_EMBD / 32;               // 32 n-tiles
  int mt = wave / NT, nt = wave % NT;
  int m0 = mt * 32, n0 = nt * 32;

  v8f acc[2][2] = {};
  for (int k0 = 0; k0 < C_EMBD; k0 += 32) {
    v16bf a0 = load_fragA(att + (size_t)m0 * C_EMBD + k0, C_EMBD, lane);
    v16bf a1 = load_fragA(att + (size_t)(m0 + 16) * C_EMBD + k0, C_EMBD, lane);
    v16bf b0 = load_fragB(wpT + (size_t)n0 * C_EMBD + k0, C_EMBD, lane);
    v16bf b1 = load_fragB(wpT + (size_t)(n0 + 16) * C_EMBD + k0, C_EMBD, lane);
    acc[0][0] = wmma_bf16(a0, b0, acc[0][0]);
    acc[0][1] = wmma_bf16(a0, b1, acc[0][1]);
    acc[1][0] = wmma_bf16(a1, b0, acc[1][0]);
    acc[1][1] = wmma_bf16(a1, b1, acc[1][1]);
  }
  for (int i = 0; i < 2; ++i)
    for (int j = 0; j < 2; ++j)
      for (int r = 0; r < 8; ++r) {
        int m = m0 + 16 * i + r + 8 * (lane >> 4);
        int n = n0 + 16 * j + (lane & 15);
        out[(size_t)m * C_EMBD + n] = acc[i][j][r] + b_proj[n];
      }
}

// ---------------- launcher ----------------
extern "C" void kernel_launch(void* const* d_in, const int* in_sizes, int n_in,
                              void* d_out, int out_size, void* d_ws, size_t ws_size,
                              hipStream_t stream) {
  (void)in_sizes; (void)n_in; (void)out_size; (void)ws_size;
  const float* x      = (const float*)d_in[0];
  const float* W_attn = (const float*)d_in[1];
  const float* b_attn = (const float*)d_in[2];
  const float* W_proj = (const float*)d_in[3];
  const float* b_proj = (const float*)d_in[4];
  float* out = (float*)d_out;

  char* ws = (char*)d_ws;
  size_t off = 0;
  auto alloc = [&](size_t bytes) -> void* {
    void* p = ws + off;
    off = (off + bytes + 255) & ~(size_t)255;
    return p;
  };
  u16* xb  = (u16*)alloc((size_t)MROWS * C_EMBD * 2);
  u16* waT = (u16*)alloc((size_t)NQKV * C_EMBD * 2);
  u16* wpT = (u16*)alloc((size_t)C_EMBD * C_EMBD * 2);
  u16* Q   = (u16*)alloc((size_t)BATCH * HEADS * TSEQ * HDIM * 2);
  u16* Kk  = (u16*)alloc((size_t)BATCH * HEADS * TSEQ * HDIM * 2);
  u16* Vt  = (u16*)alloc((size_t)BATCH * HEADS * HDIM * TSEQ * 2);
  u16* att = (u16*)alloc((size_t)MROWS * C_EMBD * 2);

  int nx = MROWS * C_EMBD;
  k_cvt<<<(nx + 255) / 256, 256, 0, stream>>>(x, xb, nx);
  k_cvt_T<<<(NQKV * C_EMBD + 255) / 256, 256, 0, stream>>>(W_attn, waT, C_EMBD, NQKV);
  k_cvt_T<<<(C_EMBD * C_EMBD + 255) / 256, 256, 0, stream>>>(W_proj, wpT, C_EMBD, C_EMBD);

  int qkv_waves  = (MROWS / 32) * (NQKV / 32);    // 12288
  k_qkv<<<qkv_waves / 8, 256, 0, stream>>>(xb, waT, b_attn, Q, Kk, Vt);

  int att_waves  = BATCH * HEADS * (TSEQ / 16);   // 4096
  k_attn<<<att_waves / 8, 256, 0, stream>>>(Q, Kk, Vt, att);

  int proj_waves = (MROWS / 32) * (C_EMBD / 32);  // 4096
  k_proj<<<proj_waves / 8, 256, 0, stream>>>(att, wpT, b_proj, out);
}
